// CIGINModel_19344532701719
// MI455X (gfx1250) — matrix operand order repaired and checked by reading, not compiled, not run on hardware
//
#include <hip/hip_runtime.h>
#include <math.h>

typedef __attribute__((ext_vector_type(16))) _Float16 v16h;
typedef __attribute__((ext_vector_type(8)))  _Float16 v8h;
typedef __attribute__((ext_vector_type(8)))  float    v8f;

#define D 42
#define DPAD 64   // padded K for f16 WMMA (42 -> 64, zeros beyond 41)

static __device__ __forceinline__ float sigmoidf_(float x) {
    return 1.0f / (1.0f + __expf(-x));
}

// ---------------------------------------------------------------------------
// h0 = relu(x @ lin0_W + lin0_b)      [N,42] @ [42,42]
// ---------------------------------------------------------------------------
__global__ void lin0_kernel(const float* __restrict__ x,
                            const float* __restrict__ W,
                            const float* __restrict__ b,
                            float* __restrict__ h, int N) {
    int gid = blockIdx.x * blockDim.x + threadIdx.x;
    if (gid >= N * D) return;
    int n = gid / D, o = gid % D;
    const float* xr = x + (size_t)n * D;
    float acc = b[o];
    #pragma unroll 7
    for (int k = 0; k < D; ++k) acc += xr[k] * W[k * D + o];
    h[gid] = fmaxf(acc, 0.0f);
}

// ---------------------------------------------------------------------------
// r = relu(e @ en1_W + en1_b)         [E,10] @ [10,10]
// ---------------------------------------------------------------------------
__global__ void edge_relu_kernel(const float* __restrict__ e,
                                 const float* __restrict__ W,
                                 const float* __restrict__ b,
                                 float* __restrict__ r, int E) {
    int gid = blockIdx.x * blockDim.x + threadIdx.x;
    if (gid >= E * 10) return;
    int ed = gid / 10, k = gid % 10;
    const float* er = e + (size_t)ed * 10;
    float acc = b[k];
    #pragma unroll
    for (int i = 0; i < 10; ++i) acc += er[i] * W[i * 10 + k];
    r[gid] = fmaxf(acc, 0.0f);
}

// ---------------------------------------------------------------------------
// G[n][k*42+o] = sum_i h[n][i] * (k<10 ? en2_W[k][i*42+o] : en2_b[i*42+o])
// One block per node; G fits in L2 (15 MB).
// ---------------------------------------------------------------------------
__global__ void node_gemm_kernel(const float* __restrict__ h,
                                 const float* __restrict__ en2_W,
                                 const float* __restrict__ en2_b,
                                 float* __restrict__ G) {
    __shared__ float hrow[D];
    int n = blockIdx.x;
    int t = threadIdx.x;
    if (t < D) hrow[t] = h[(size_t)n * D + t];
    __syncthreads();
    if (t >= 11 * D) return;
    int k = t / D, o = t % D;
    const float* Wp = (k < 10) ? (en2_W + (size_t)k * (D * D)) : en2_b;
    float acc = 0.0f;
    #pragma unroll 7
    for (int i = 0; i < D; ++i) acc += hrow[i] * Wp[i * D + o];
    G[(size_t)n * (11 * D) + t] = acc;
}

// ---------------------------------------------------------------------------
// per-edge message + scatter-add into agg
// ---------------------------------------------------------------------------
__global__ void edge_msg_kernel(const float* __restrict__ r,
                                const float* __restrict__ G,
                                const int* __restrict__ src,
                                const int* __restrict__ dst,
                                float* __restrict__ agg, int E) {
    int gid = blockIdx.x * blockDim.x + threadIdx.x;
    if (gid >= E * D) return;
    int e = gid / D, o = gid % D;
    const float* Gb = G + (size_t)src[e] * (11 * D);
    const float* rr = r + (size_t)e * 10;
    float acc = Gb[10 * D + o];           // bias-matrix term
    #pragma unroll
    for (int k = 0; k < 10; ++k) acc += rr[k] * Gb[k * D + o];
    atomicAdd(agg + (size_t)dst[e] * D + o, acc);
}

// ---------------------------------------------------------------------------
// h_new = [relu(agg+h), h] @ msg_W + msg_b
// ---------------------------------------------------------------------------
__global__ void combine_kernel(const float* __restrict__ agg,
                               const float* __restrict__ h,
                               const float* __restrict__ W,   // [84,42]
                               const float* __restrict__ b,
                               float* __restrict__ hn, int N) {
    int gid = blockIdx.x * blockDim.x + threadIdx.x;
    if (gid >= N * D) return;
    int n = gid / D, o = gid % D;
    const float* ar = agg + (size_t)n * D;
    const float* hr = h + (size_t)n * D;
    float acc = b[o];
    #pragma unroll 7
    for (int i = 0; i < D; ++i) {
        float m = fmaxf(ar[i] + hr[i], 0.0f);
        acc += m * W[i * D + o] + hr[i] * W[(D + i) * D + o];
    }
    hn[gid] = acc;
}

__global__ void add_x_kernel(const float* __restrict__ h,
                             const float* __restrict__ x,
                             float* __restrict__ out, int n) {
    int gid = blockIdx.x * blockDim.x + threadIdx.x;
    if (gid < n) out[gid] = h[gid] + x[gid];
}

__global__ void zero_f32_kernel(float* __restrict__ p, int n) {
    int gid = blockIdx.x * blockDim.x + threadIdx.x;
    if (gid < n) p[gid] = 0.0f;
}

// ---------------------------------------------------------------------------
// Pack h [N,42] f32 -> hf16 [N,64] (zero padded) and hT16 [64,N]
// ---------------------------------------------------------------------------
__global__ void pack_f16_kernel(const float* __restrict__ h,
                                _Float16* __restrict__ hf,
                                _Float16* __restrict__ hT, int N) {
    int gid = blockIdx.x * blockDim.x + threadIdx.x;
    if (gid >= N * DPAD) return;
    int n = gid / DPAD, f = gid % DPAD;
    float v = (f < D) ? h[(size_t)n * D + f] : 0.0f;
    _Float16 hv = (_Float16)v;
    hf[(size_t)n * DPAD + f] = hv;
    hT[(size_t)f * N + n]    = hv;
}

// ---------------------------------------------------------------------------
// Interaction map: I = h_su @ h_sv^T   (f16 in / f32 acc via WMMA)
// one wave per 16x16 tile; K = 64 (padded) -> 2 WMMAs
// ---------------------------------------------------------------------------
__global__ void interaction_wmma_kernel(const _Float16* __restrict__ Af,
                                        const _Float16* __restrict__ Bf,
                                        float* __restrict__ I,
                                        int N, int tilesN) {
    int wave = threadIdx.x >> 5;
    int lane = threadIdx.x & 31;
    int tile = blockIdx.x * (blockDim.x >> 5) + wave;
    int tileM = tile / tilesN;
    int tileN = tile % tilesN;
    if (tileM >= tilesN) return;
    int m = lane & 15, half = lane >> 4;

    const _Float16* ap = Af + (size_t)(tileM * 16 + m) * DPAD + half * 16;
    const _Float16* bp = Bf + (size_t)(tileN * 16 + m) * DPAD + half * 16;

    v8f c = {};
    #pragma unroll
    for (int kc = 0; kc < DPAD; kc += 32) {
        v8h alo = *(const v8h*)(ap + kc);
        v8h ahi = *(const v8h*)(ap + kc + 8);
        v16h a = __builtin_shufflevector(alo, ahi, 0, 1, 2, 3, 4, 5, 6, 7,
                                         8, 9, 10, 11, 12, 13, 14, 15);
        v16h b = *(const v16h*)(bp + kc);
        c = __builtin_amdgcn_wmma_f32_16x16x32_f16(false, a, false, b,
                                                   (short)0, c, false, false);
    }
    float* out = I + (size_t)(tileM * 16 + half * 8) * N + tileN * 16 + m;
    #pragma unroll
    for (int v = 0; v < 8; ++v) out[(size_t)v * N] = c[v];
}

// ---------------------------------------------------------------------------
// solute_prime = tanh(I) @ h_sv   : out [M,42], K = Nsv, BT = hT_sv [64,K]
// tile encodes (row-tile, k-segment of 8); 3 N-tiles per wave; atomic acc.
// ---------------------------------------------------------------------------
__global__ void solute_prime_wmma_kernel(const float* __restrict__ I,
                                         const _Float16* __restrict__ BT,
                                         float* __restrict__ out,
                                         int M, int K) {
    int wave = threadIdx.x >> 5;
    int lane = threadIdx.x & 31;
    int tile = blockIdx.x * (blockDim.x >> 5) + wave;
    int tileM = tile >> 3, kseg = tile & 7;
    if (tileM >= M / 16) return;
    int m = lane & 15, half = lane >> 4, kb = half * 16;

    const float* arow = I + (size_t)(tileM * 16 + m) * K;
    const _Float16* bp0 = BT + (size_t)(0 + m) * K + half * 16;
    const _Float16* bp1 = BT + (size_t)(16 + m) * K + half * 16;
    const _Float16* bp2 = BT + (size_t)(32 + m) * K + half * 16;

    v8f c0 = {}, c1 = {}, c2 = {};
    int k0 = kseg * (K / 8), k1 = k0 + K / 8;
    for (int kc = k0; kc < k1; kc += 32) {
        __builtin_prefetch(arow + kc + 256, 0, 0);
        v16h a;
        #pragma unroll
        for (int j = 0; j < 8; ++j) {
            a[j]     = (_Float16)tanhf(arow[kc + kb + j]);
            a[8 + j] = (_Float16)tanhf(arow[kc + kb + 8 + j]);
        }
        v16h b0 = *(const v16h*)(bp0 + kc);
        v16h b1 = *(const v16h*)(bp1 + kc);
        v16h b2 = *(const v16h*)(bp2 + kc);
        c0 = __builtin_amdgcn_wmma_f32_16x16x32_f16(false, a, false, b0, (short)0, c0, false, false);
        c1 = __builtin_amdgcn_wmma_f32_16x16x32_f16(false, a, false, b1, (short)0, c1, false, false);
        c2 = __builtin_amdgcn_wmma_f32_16x16x32_f16(false, a, false, b2, (short)0, c2, false, false);
    }
    int row0 = tileM * 16 + half * 8;
    #pragma unroll
    for (int v = 0; v < 8; ++v) {
        float* orow = out + (size_t)(row0 + v) * D;
        atomicAdd(orow + m, c0[v]);
        atomicAdd(orow + 16 + m, c1[v]);
        if (32 + m < D) atomicAdd(orow + 32 + m, c2[v]);
    }
}

// ---------------------------------------------------------------------------
// solvent_prime = tanh(I)^T @ h_su : out [Msv,42], K = Nsu, I is [K, ldI]
// ---------------------------------------------------------------------------
__global__ void solvent_prime_wmma_kernel(const float* __restrict__ I,
                                          const _Float16* __restrict__ BT,
                                          float* __restrict__ out,
                                          int M, int K, int ldI) {
    int wave = threadIdx.x >> 5;
    int lane = threadIdx.x & 31;
    int tile = blockIdx.x * (blockDim.x >> 5) + wave;
    int tileM = tile >> 3, kseg = tile & 7;
    if (tileM >= M / 16) return;
    int m = lane & 15, half = lane >> 4, kb = half * 16;

    int col = tileM * 16 + m;
    const _Float16* bp0 = BT + (size_t)(0 + m) * K + half * 16;
    const _Float16* bp1 = BT + (size_t)(16 + m) * K + half * 16;
    const _Float16* bp2 = BT + (size_t)(32 + m) * K + half * 16;

    v8f c0 = {}, c1 = {}, c2 = {};
    int k0 = kseg * (K / 8), k1 = k0 + K / 8;
    for (int kc = k0; kc < k1; kc += 32) {
        __builtin_prefetch(I + (size_t)(kc + 32) * ldI + col, 0, 0);
        v16h a;
        #pragma unroll
        for (int j = 0; j < 8; ++j) {
            a[j]     = (_Float16)tanhf(I[(size_t)(kc + kb + j) * ldI + col]);
            a[8 + j] = (_Float16)tanhf(I[(size_t)(kc + kb + 8 + j) * ldI + col]);
        }
        v16h b0 = *(const v16h*)(bp0 + kc);
        v16h b1 = *(const v16h*)(bp1 + kc);
        v16h b2 = *(const v16h*)(bp2 + kc);
        c0 = __builtin_amdgcn_wmma_f32_16x16x32_f16(false, a, false, b0, (short)0, c0, false, false);
        c1 = __builtin_amdgcn_wmma_f32_16x16x32_f16(false, a, false, b1, (short)0, c1, false, false);
        c2 = __builtin_amdgcn_wmma_f32_16x16x32_f16(false, a, false, b2, (short)0, c2, false, false);
    }
    int row0 = tileM * 16 + half * 8;
    #pragma unroll
    for (int v = 0; v < 8; ++v) {
        float* orow = out + (size_t)(row0 + v) * D;
        atomicAdd(orow + m, c0[v]);
        atomicAdd(orow + 16 + m, c1[v]);
        if (32 + m < D) atomicAdd(orow + 32 + m, c2[v]);
    }
}

// ---------------------------------------------------------------------------
// Set2Set: single block, 2 LSTM+attention iterations, feat = [h | prime]
// ---------------------------------------------------------------------------
__global__ void set2set_kernel(const float* __restrict__ h,
                               const float* __restrict__ prime,
                               const float* __restrict__ Wih,  // [168,336]
                               const float* __restrict__ Whh,  // [84,336]
                               const float* __restrict__ bih,
                               const float* __restrict__ bhh,
                               float* __restrict__ qout, int N) {
    __shared__ float qstar[168], hhv[84], ccv[84], gates[336], red[512];
    __shared__ float logits[8192];
    __shared__ float smax, ssum;
    int t = threadIdx.x;
    if (t < 168) qstar[t] = 0.0f;
    if (t < 84) { hhv[t] = 0.0f; ccv[t] = 0.0f; }
    __syncthreads();

    for (int it = 0; it < 2; ++it) {
        if (t < 336) {
            float g = bih[t] + bhh[t];
            for (int j = 0; j < 168; ++j) g += qstar[j] * Wih[j * 336 + t];
            for (int j = 0; j < 84; ++j)  g += hhv[j] * Whh[j * 336 + t];
            gates[t] = g;
        }
        __syncthreads();
        if (t < 84) {
            float ig = sigmoidf_(gates[t]);
            float fg = sigmoidf_(gates[84 + t]);
            float gg = tanhf(gates[168 + t]);
            float og = sigmoidf_(gates[252 + t]);
            float c2 = fg * ccv[t] + ig * gg;
            ccv[t] = c2;
            hhv[t] = og * tanhf(c2);
        }
        __syncthreads();
        for (int n = t; n < N; n += blockDim.x) {
            const float* hr = h + (size_t)n * D;
            const float* pr = prime + (size_t)n * D;
            float s = 0.0f;
            #pragma unroll 7
            for (int j = 0; j < D; ++j) s += hr[j] * hhv[j] + pr[j] * hhv[D + j];
            logits[n] = s;
        }
        __syncthreads();
        float lm = -3.4e38f;
        for (int n = t; n < N; n += blockDim.x) lm = fmaxf(lm, logits[n]);
        red[t] = lm; __syncthreads();
        for (int s = 256; s > 0; s >>= 1) {
            if (t < s) red[t] = fmaxf(red[t], red[t + s]);
            __syncthreads();
        }
        if (t == 0) smax = red[0];
        __syncthreads();
        float ls = 0.0f;
        for (int n = t; n < N; n += blockDim.x) {
            float e = __expf(logits[n] - smax);
            logits[n] = e;
            ls += e;
        }
        red[t] = ls; __syncthreads();
        for (int s = 256; s > 0; s >>= 1) {
            if (t < s) red[t] += red[t + s];
            __syncthreads();
        }
        if (t == 0) ssum = red[0];
        __syncthreads();
        if (t < 84) {
            float rv = 0.0f;
            for (int n = 0; n < N; ++n) {
                float fv = (t < D) ? h[(size_t)n * D + t]
                                   : prime[(size_t)n * D + (t - D)];
                rv += logits[n] * fv;
            }
            rv /= ssum;
            qstar[t] = hhv[t];
            qstar[84 + t] = rv;
        }
        __syncthreads();
    }
    if (t < 168) qout[t] = qstar[t];
}

// ---------------------------------------------------------------------------
// Final MLP: [q_su|q_sv](336) -> 256 relu -> 128 relu -> 1
// ---------------------------------------------------------------------------
__global__ void final_fc_kernel(const float* __restrict__ qsu,
                                const float* __restrict__ qsv,
                                const float* __restrict__ W1, const float* __restrict__ b1,
                                const float* __restrict__ W2, const float* __restrict__ b2,
                                const float* __restrict__ W3, const float* __restrict__ b3,
                                float* __restrict__ out0) {
    __shared__ float fin[336], p1[256], p2[128], red[128];
    int t = threadIdx.x;
    if (t < 168) { fin[t] = qsu[t]; fin[168 + t] = qsv[t]; }
    __syncthreads();
    float v = b1[t];
    for (int j = 0; j < 336; ++j) v += fin[j] * W1[j * 256 + t];
    p1[t] = fmaxf(v, 0.0f);
    __syncthreads();
    if (t < 128) {
        float u = b2[t];
        for (int j = 0; j < 256; ++j) u += p1[j] * W2[j * 128 + t];
        p2[t] = fmaxf(u, 0.0f);
    }
    __syncthreads();
    if (t < 128) red[t] = p2[t] * W3[t];
    __syncthreads();
    for (int s = 64; s > 0; s >>= 1) {
        if (t < s) red[t] += red[t + s];
        __syncthreads();
    }
    if (t == 0) out0[0] = red[0] + b3[0];
}

// ===========================================================================
extern "C" void kernel_launch(void* const* d_in, const int* in_sizes, int n_in,
                              void* d_out, int out_size, void* d_ws, size_t ws_size,
                              hipStream_t stream) {
    (void)n_in; (void)out_size; (void)ws_size;
    const int N = in_sizes[0] / D;     // 8192
    const int E = in_sizes[2];         // 32768

    // ---- workspace bump allocator (256B aligned) ----
    size_t off = 0;
    auto alloc = [&](size_t bytes) -> void* {
        off = (off + 255) & ~(size_t)255;
        void* p = (char*)d_ws + off;
        off += bytes;
        return p;
    };
    float* hA[2];  float* hB[2];  float* rb[2];
    float* prime[2]; float* qv[2];
    _Float16* hf16[2]; _Float16* hT16[2];
    for (int b = 0; b < 2; ++b) {
        hA[b]    = (float*)alloc((size_t)N * D * 4);
        hB[b]    = (float*)alloc((size_t)N * D * 4);
        rb[b]    = (float*)alloc((size_t)E * 10 * 4);
        prime[b] = (float*)alloc((size_t)N * D * 4);
        qv[b]    = (float*)alloc(168 * 4);
        hf16[b]  = (_Float16*)alloc((size_t)N * DPAD * 2);
        hT16[b]  = (_Float16*)alloc((size_t)DPAD * N * 2);
    }
    float* G   = (float*)alloc((size_t)N * 11 * D * 4);   // 15.1 MB, L2-resident
    float* agg = (float*)alloc((size_t)N * D * 4);

    float* I = (float*)d_out + 1;   // interaction map [N,N] (output #2)
    float* pred = (float*)d_out;    // prediction scalar (output #1)

    const int ND = N * D;
    float* hfin[2];

    // -------- MPNN branches --------
    for (int b = 0; b < 2; ++b) {
        const float* x      = (const float*)d_in[b * 16 + 0];
        const float* e      = (const float*)d_in[b * 16 + 1];
        const int*   src    = (const int*)d_in[b * 16 + 2];
        const int*   dst    = (const int*)d_in[b * 16 + 3];
        const float* lin0_W = (const float*)d_in[b * 16 + 4];
        const float* lin0_b = (const float*)d_in[b * 16 + 5];
        const float* en1_W  = (const float*)d_in[b * 16 + 6];
        const float* en1_b  = (const float*)d_in[b * 16 + 7];
        const float* en2_W  = (const float*)d_in[b * 16 + 8];
        const float* en2_b  = (const float*)d_in[b * 16 + 9];
        const float* msg_W  = (const float*)d_in[b * 16 + 10];
        const float* msg_b  = (const float*)d_in[b * 16 + 11];

        lin0_kernel<<<(ND + 255) / 256, 256, 0, stream>>>(x, lin0_W, lin0_b, hA[b], N);
        edge_relu_kernel<<<(E * 10 + 255) / 256, 256, 0, stream>>>(e, en1_W, en1_b, rb[b], E);

        float* hc = hA[b];
        float* hn = hB[b];
        for (int step = 0; step < 8; ++step) {
            node_gemm_kernel<<<N, 512, 0, stream>>>(hc, en2_W, en2_b, G);
            zero_f32_kernel<<<(ND + 255) / 256, 256, 0, stream>>>(agg, ND);
            edge_msg_kernel<<<(E * D + 255) / 256, 256, 0, stream>>>(rb[b], G, src, dst, agg, E);
            combine_kernel<<<(ND + 255) / 256, 256, 0, stream>>>(agg, hc, msg_W, msg_b, hn, N);
            float* tmp = hc; hc = hn; hn = tmp;
        }
        add_x_kernel<<<(ND + 255) / 256, 256, 0, stream>>>(hc, x, hn, ND);
        hfin[b] = hn;
        pack_f16_kernel<<<(N * DPAD + 255) / 256, 256, 0, stream>>>(hn, hf16[b], hT16[b], N);
    }

    // -------- Interaction map: I = h_su @ h_sv^T (WMMA) --------
    {
        int tilesN = N / 16;
        int tiles = tilesN * tilesN;
        interaction_wmma_kernel<<<tiles / 8, 256, 0, stream>>>(hf16[0], hf16[1], I, N, tilesN);
    }

    // -------- Projections: solute' = tanh(I) @ h_sv ; solvent' = tanh(I)^T @ h_su --------
    zero_f32_kernel<<<(ND + 255) / 256, 256, 0, stream>>>(prime[0], ND);
    zero_f32_kernel<<<(ND + 255) / 256, 256, 0, stream>>>(prime[1], ND);
    {
        int tiles = (N / 16) * 8;   // (row-tile, k-segment)
        solute_prime_wmma_kernel<<<tiles / 8, 256, 0, stream>>>(I, hT16[1], prime[0], N, N);
        solvent_prime_wmma_kernel<<<tiles / 8, 256, 0, stream>>>(I, hT16[0], prime[1], N, N, N);
    }

    // -------- Set2Set per branch --------
    for (int b = 0; b < 2; ++b) {
        const float* Wih = (const float*)d_in[b * 16 + 12];
        const float* Whh = (const float*)d_in[b * 16 + 13];
        const float* bih = (const float*)d_in[b * 16 + 14];
        const float* bhh = (const float*)d_in[b * 16 + 15];
        set2set_kernel<<<1, 512, 0, stream>>>(hfin[b], prime[b], Wih, Whh, bih, bhh, qv[b], N);
    }

    // -------- Final MLP --------
    final_fc_kernel<<<1, 256, 0, stream>>>(
        qv[0], qv[1],
        (const float*)d_in[32], (const float*)d_in[33],
        (const float*)d_in[34], (const float*)d_in[35],
        (const float*)d_in[36], (const float*)d_in[37],
        pred);
}
